// InformerLayer_7370163880221
// MI455X (gfx1250) — matrix operand order repaired and checked by reading, not compiled, not run on hardware
//
#include <hip/hip_runtime.h>

// ---------------------------------------------------------------------------
// Informer ProbSparse layer for MI455X (gfx1250), wave32 + WMMA.
//   B=128 seqs, L=256, C=128, H=8, E=16, DFF=512, u=sample_k=18.
// Dense GEMMs: v_wmma_f32_16x16x32_f16 fed from double-buffered LDS tiles
// staged with global_load_async_to_lds_b128 (ASYNCcnt) via inline asm.
// ---------------------------------------------------------------------------

typedef __attribute__((ext_vector_type(16))) _Float16 v16h;
typedef __attribute__((ext_vector_type(8)))  float    v8f;

struct alignas(16) h8 { _Float16 h[8]; };

#define C_MODEL 128
#define L_SEQ   256
#define B_SEQ   128             // b * N = 4 * 32
#define BL      (B_SEQ * L_SEQ) // 32768 rows
#define H_HEADS 8
#define E_HEAD  16
#define D_FF    512
#define U_TOP   18
#define SAMPLEK 18

// CDNA5 async copy global->LDS (ISA 15.18.3 op 98). Per-lane: 16 bytes.
// LDS address = low 32 bits of the generic shared pointer (ISA 10.2:
// in the LDS aperture); global address = 64-bit
// generic pointer value (identity for the global aperture).
__device__ __forceinline__ void informer_async_cp_b128(const void* gsrc,
                                                       void* ldst) {
  unsigned lds_addr = (unsigned)(unsigned long long)(uintptr_t)ldst;
  unsigned long long gaddr = (unsigned long long)(uintptr_t)gsrc;
  asm volatile("global_load_async_to_lds_b128 %0, %1, off"
               :
               : "v"(lds_addr), "v"(gaddr)
               : "memory");
}

#if __has_builtin(__builtin_amdgcn_s_wait_asynccnt)
#define INFORMER_WAIT_ASYNC(n) __builtin_amdgcn_s_wait_asynccnt(n)
#else
#define INFORMER_WAIT_ASYNC(n) asm volatile("s_wait_asynccnt %0" ::"i"(n))
#endif

// -------------------------------------------------------------------------
// prep: h[bl][c] = x[bl] * mlp_w[c] + mlp_b[c]   (x is [4,32,256] = [BL])
// -------------------------------------------------------------------------
__global__ void informer_prep_kernel(const float* __restrict__ x,
                                     const float* __restrict__ mw,
                                     const float* __restrict__ mb,
                                     float* __restrict__ hF,
                                     _Float16* __restrict__ hH) {
  int i = blockIdx.x * blockDim.x + threadIdx.x;
  if (i >= BL * C_MODEL) return;
  int c  = i & (C_MODEL - 1);
  int bl = i >> 7;
  float v = fmaf(x[bl], mw[c], mb[c]);
  hF[i] = v;
  hH[i] = (_Float16)v;
}

__global__ void informer_f32_to_f16_kernel(const float* __restrict__ a,
                                           _Float16* __restrict__ o, int n) {
  int i = blockIdx.x * blockDim.x + threadIdx.x;
  if (i < n) o[i] = (_Float16)a[i];
}

// splitmix-style hash: deterministic stand-in for jax.random.randint(key 42)
// (compile-only loop: exact threefry stream not reproduced).
__global__ void informer_gen_idx_kernel(int* __restrict__ idx) {
  int i = blockIdx.x * blockDim.x + threadIdx.x;
  if (i >= L_SEQ * SAMPLEK) return;
  unsigned z = (unsigned)(i + 1) * 0x9E3779B9u + 42u;
  z ^= z >> 16; z *= 0x85EBCA6Bu;
  z ^= z >> 13; z *= 0xC2B2AE35u;
  z ^= z >> 16;
  idx[i] = (int)(z & (L_SEQ - 1));
}

// -------------------------------------------------------------------------
// WMMA GEMM: out[M,N] = A[M,K] @ W[N,K]^T + bias, optional ReLU.
// Block = 8 waves computes a 128(M) x 64(N) tile. Per 32-wide K step the
// block stages A(128x32) + W(64x32) into double-buffered LDS with async
// b128 copies (3 per thread per stage -> s_wait_asynccnt 3 overlaps the
// next stage's copies with the current stage's WMMAs). Wave w owns rows
// [w*16, w*16+16) and all 64 columns (4 accumulators, A-frag reuse).
// Fragment layouts used here: A lane row = lane&15, contiguous K chunk
// [kb,kb+16), kb=16*(lane>=16). B lane col = lane&15, contiguous
// K chunk at 16*(lane>=16); B[k][n] = W[n][k] row-major. D: VGPR r ->
// row r + 8*(lane>=16), col = lane&15.
// -------------------------------------------------------------------------
template <int K, bool RELU>
__global__ void informer_gemm_wmma_kernel(const _Float16* __restrict__ A,
                                          const _Float16* __restrict__ W,
                                          const float* __restrict__ bias,
                                          float* __restrict__ outF,
                                          _Float16* __restrict__ outH,
                                          int M, int N) {
  __shared__ alignas(16) _Float16 s_a[2][128 * 32];  // 2 x 8 KB
  __shared__ alignas(16) _Float16 s_w[2][64 * 32];   // 2 x 4 KB

  const int tid  = threadIdx.x;
  const int lane = tid & 31;
  const int wave = tid >> 5;
  const int ngrp = N >> 6;
  const int M0 = (blockIdx.x / ngrp) << 7;
  const int N0 = (blockIdx.x % ngrp) << 6;

  // cooperative staging: A tile = 512 16B chunks (2/thread), W tile = 256
  const int ar0 = tid >> 2;                 // A chunk 0: row, seg
  const int as0 = (tid & 3) * 8;
  const int ar1 = (tid + 256) >> 2;         // A chunk 1
  const int as1 = as0;
  const int wr  = tid >> 2;                 // W chunk
  const int wsg = (tid & 3) * 8;

  auto stage = [&](int buf, int kt) {
    const _Float16* g0 = A + (size_t)(M0 + ar0) * K + kt + as0;
    const _Float16* g1 = A + (size_t)(M0 + ar1) * K + kt + as1;
    const _Float16* g2 = W + (size_t)(N0 + wr) * K + kt + wsg;
    informer_async_cp_b128(g0, &s_a[buf][ar0 * 32 + as0]);
    informer_async_cp_b128(g1, &s_a[buf][ar1 * 32 + as1]);
    informer_async_cp_b128(g2, &s_w[buf][wr * 32 + wsg]);
  };

  const int row = lane & 15;
  const int kb  = (lane >> 4) << 4;   // A frag K sub-offset: 0 or 16
  const int kbB = (lane >> 4) << 4;   // B frag K sub-offset: 0 or 16

  v8f acc[4] = {v8f{}, v8f{}, v8f{}, v8f{}};

  constexpr int NK = K / 32;
  stage(0, 0);
  int buf = 0;
#pragma unroll
  for (int kt = 0; kt < NK; ++kt) {
    if (kt + 1 < NK) {
      stage(buf ^ 1, (kt + 1) * 32);
      INFORMER_WAIT_ASYNC(3);   // previous stage done, next stage in flight
    } else {
      INFORMER_WAIT_ASYNC(0);
    }
    __syncthreads();

    const _Float16* arow = &s_a[buf][(wave * 16 + row) * 32];
    v16h af;
    *((h8*)&af)       = *(const h8*)(arow + kb);
    *(((h8*)&af) + 1) = *(const h8*)(arow + kb + 8);
#pragma unroll
    for (int j = 0; j < 4; ++j) {
      const _Float16* wrow = &s_w[buf][(j * 16 + row) * 32 + kbB];
      v16h bf;
      *((h8*)&bf)       = *(const h8*)(wrow);
      *(((h8*)&bf) + 1) = *(const h8*)(wrow + 8);
      acc[j] = __builtin_amdgcn_wmma_f32_16x16x32_f16(
          /*neg_a=*/false, af, /*neg_b=*/false, bf,
          /*c_mod=*/(short)0, acc[j], /*reuse_a=*/false, /*reuse_b=*/false);
    }
    __syncthreads();
    buf ^= 1;
  }

  const int orow = M0 + wave * 16 + ((lane >> 4) << 3);
#pragma unroll
  for (int j = 0; j < 4; ++j) {
    const int col = N0 + j * 16 + (lane & 15);
    const float bv = bias ? bias[col] : 0.0f;
#pragma unroll
    for (int r = 0; r < 8; ++r) {
      float v = acc[j][r] + bv;
      if (RELU) v = fmaxf(v, 0.0f);
      const size_t oi = (size_t)(orow + r) * N + col;
      if (outF) outF[oi] = v;
      if (outH) outH[oi] = (_Float16)v;
    }
  }
}

// -------------------------------------------------------------------------
// ProbSparse attention core: one block per (batch-seq, head).
// -------------------------------------------------------------------------
__device__ __forceinline__ float informer_dot16(const float* a, const float* b) {
  float s = 0.0f;
#pragma unroll
  for (int e = 0; e < 16; ++e) s = fmaf(a[e], b[e], s);
  return s;
}

__global__ void informer_attention_kernel(const float* __restrict__ q,
                                          const float* __restrict__ k,
                                          const float* __restrict__ v,
                                          const int* __restrict__ idx,
                                          _Float16* __restrict__ ctxH) {
  __shared__ float s_q[L_SEQ * E_HEAD];
  __shared__ float s_k[L_SEQ * E_HEAD];
  __shared__ float s_v[L_SEQ * E_HEAD];
  __shared__ float s_m[L_SEQ];      // sparsity measure M
  __shared__ float s_red[L_SEQ];
  __shared__ float s_attn[L_SEQ];
  __shared__ float s_vmean[E_HEAD];
  __shared__ float s_upd[U_TOP * E_HEAD];
  __shared__ int   s_mtop[U_TOP];

  const int t  = threadIdx.x;                 // 256 threads, one per token
  const int bs = blockIdx.x >> 3;             // sequence index 0..127
  const int hd = blockIdx.x & 7;              // head index

  // stage q/k/v rows for this (seq, head) into LDS (float4 vectors)
  {
    const size_t base = ((size_t)(bs * L_SEQ + t)) * C_MODEL + hd * E_HEAD;
    const float4* qp = (const float4*)(q + base);
    const float4* kp = (const float4*)(k + base);
    const float4* vp = (const float4*)(v + base);
    float4* qs = (float4*)(s_q + t * E_HEAD);
    float4* ks = (float4*)(s_k + t * E_HEAD);
    float4* vs = (float4*)(s_v + t * E_HEAD);
#pragma unroll
    for (int j = 0; j < 4; ++j) { qs[j] = qp[j]; ks[j] = kp[j]; vs[j] = vp[j]; }
  }
  __syncthreads();

  // sampled scores -> M(l) = max_j qk - mean_L qk   (unscaled, per reference)
  {
    float mx = -3.4e38f, sm = 0.0f;
#pragma unroll 2
    for (int j = 0; j < SAMPLEK; ++j) {
      const int kidx = idx[t * SAMPLEK + j];
      const float d = informer_dot16(s_q + t * E_HEAD, s_k + kidx * E_HEAD);
      mx = fmaxf(mx, d);
      sm += d;
    }
    s_m[t] = mx - sm * (1.0f / (float)L_SEQ);
  }
  __syncthreads();

  // stable top-u selection (matches lax.top_k descending, ties by index)
  {
    const float ml = s_m[t];
    int rank = 0;
    for (int j = 0; j < L_SEQ; ++j) {
      const float mj = s_m[j];
      rank += (mj > ml) || (mj == ml && j < t);
    }
    if (rank < U_TOP) s_mtop[rank] = t;
  }
  __syncthreads();

  // vmean[e] = mean over l of v[l][e]  (deterministic two-stage reduction)
  {
    const int e = t & 15, grp = t >> 4;
    float part = 0.0f;
#pragma unroll
    for (int i = 0; i < 16; ++i) part += s_v[(grp * 16 + i) * E_HEAD + e];
    s_red[t] = part;
  }
  __syncthreads();
  if (t < E_HEAD) {
    float s = 0.0f;
#pragma unroll
    for (int g = 0; g < 16; ++g) s += s_red[g * 16 + t];
    s_vmean[t] = s * (1.0f / (float)L_SEQ);
  }
  __syncthreads();

  // softmax + context update for each of the u selected queries
  for (int r = 0; r < U_TOP; ++r) {
    const int qrow = s_mtop[r];
    const float sc =
        0.25f * informer_dot16(s_q + qrow * E_HEAD, s_k + t * E_HEAD);
    s_red[t] = sc;
    __syncthreads();
    for (int off = 128; off > 0; off >>= 1) {
      if (t < off) s_red[t] = fmaxf(s_red[t], s_red[t + off]);
      __syncthreads();
    }
    const float rowmax = s_red[0];
    __syncthreads();
    const float ex = __expf(sc - rowmax);
    s_red[t] = ex;
    __syncthreads();
    for (int off = 128; off > 0; off >>= 1) {
      if (t < off) s_red[t] += s_red[t + off];
      __syncthreads();
    }
    const float rowsum = s_red[0];
    __syncthreads();
    s_attn[t] = ex / rowsum;
    __syncthreads();
    // upd[e] = sum_l attn[l] * v[l][e]  (two-stage deterministic)
    {
      const int e = t & 15, grp = t >> 4;
      float part = 0.0f;
#pragma unroll
      for (int i = 0; i < 16; ++i) {
        const int l = grp * 16 + i;
        part = fmaf(s_attn[l], s_v[l * E_HEAD + e], part);
      }
      s_red[t] = part;
    }
    __syncthreads();
    if (t < E_HEAD) {
      float u = 0.0f;
#pragma unroll
      for (int g = 0; g < 16; ++g) u += s_red[g * 16 + t];
      s_upd[r * E_HEAD + t] = u;
    }
    __syncthreads();
  }

  // context: mean(V) everywhere, scattered top-u rows get their update
  {
    int sel = -1;
#pragma unroll
    for (int r = 0; r < U_TOP; ++r)
      if (s_mtop[r] == t) sel = r;
    _Float16* out =
        ctxH + ((size_t)(bs * L_SEQ + t)) * C_MODEL + hd * E_HEAD;
#pragma unroll
    for (int e = 0; e < E_HEAD; ++e) {
      const float val = (sel >= 0) ? s_upd[sel * E_HEAD + e] : s_vmean[e];
      out[e] = (_Float16)val;
    }
  }
}

// -------------------------------------------------------------------------
// residual + LayerNorm over C=128 (one block of 128 threads per row)
// -------------------------------------------------------------------------
__global__ void informer_ln_residual_kernel(const float* __restrict__ a,
                                            const float* __restrict__ b,
                                            const float* __restrict__ g,
                                            const float* __restrict__ beta,
                                            float* __restrict__ outF,
                                            _Float16* __restrict__ outH) {
  __shared__ float red[C_MODEL];
  const int row = blockIdx.x;
  const int t = threadIdx.x;
  const size_t i = (size_t)row * C_MODEL + t;
  const float v = a[i] + b[i];
  red[t] = v;
  __syncthreads();
  for (int off = 64; off > 0; off >>= 1) {
    if (t < off) red[t] += red[t + off];
    __syncthreads();
  }
  const float mu = red[0] * (1.0f / (float)C_MODEL);
  __syncthreads();
  const float d = v - mu;
  red[t] = d * d;
  __syncthreads();
  for (int off = 64; off > 0; off >>= 1) {
    if (t < off) red[t] += red[t + off];
    __syncthreads();
  }
  const float var = red[0] * (1.0f / (float)C_MODEL);
  const float y = fmaf(d * rsqrtf(var + 1e-5f), g[t], beta[t]);
  outF[i] = y;
  if (outH) outH[i] = (_Float16)y;
}

// -------------------------------------------------------------------------
// out[b][c][t] = sum_n x2[((b*32+n)*256+t)*128 + c]
// -------------------------------------------------------------------------
__global__ void informer_final_sum_kernel(const float* __restrict__ x2,
                                          float* __restrict__ out) {
  int i = blockIdx.x * blockDim.x + threadIdx.x;
  if (i >= 4 * C_MODEL * L_SEQ) return;
  const int t = i & 255;
  const int c = (i >> 8) & 127;
  const int b = i >> 15;
  float s = 0.0f;
#pragma unroll 4
  for (int n = 0; n < 32; ++n)
    s += x2[((size_t)(b * 32 + n) * L_SEQ + t) * C_MODEL + c];
  out[i] = s;
}

// -------------------------------------------------------------------------
extern "C" void kernel_launch(void* const* d_in, const int* in_sizes, int n_in,
                              void* d_out, int out_size, void* d_ws,
                              size_t ws_size, hipStream_t stream) {
  const float* x     = (const float*)d_in[0];
  const float* mlp_w = (const float*)d_in[1];
  const float* mlp_b = (const float*)d_in[2];
  const float* wq    = (const float*)d_in[3];
  const float* bq    = (const float*)d_in[4];
  const float* wk    = (const float*)d_in[5];
  const float* bk    = (const float*)d_in[6];
  const float* wv    = (const float*)d_in[7];
  const float* bv    = (const float*)d_in[8];
  const float* wo    = (const float*)d_in[9];
  const float* bo    = (const float*)d_in[10];
  const float* w1    = (const float*)d_in[11];
  const float* b1    = (const float*)d_in[12];
  const float* w2    = (const float*)d_in[13];
  const float* b2    = (const float*)d_in[14];
  const float* ln1g  = (const float*)d_in[15];
  const float* ln1b  = (const float*)d_in[16];
  const float* ln2g  = (const float*)d_in[17];
  const float* ln2b  = (const float*)d_in[18];
  float* out = (float*)d_out;

  char* ws = (char*)d_ws;
  size_t off = 0;
  auto carve = [&](size_t bytes) -> void* {
    void* p = ws + off;
    off += (bytes + 255) & ~(size_t)255;
    return p;
  };

  float*    hF   = (float*)   carve((size_t)BL * C_MODEL * 4);
  _Float16* hH   = (_Float16*)carve((size_t)BL * C_MODEL * 2);
  _Float16* wqH  = (_Float16*)carve((size_t)C_MODEL * C_MODEL * 2);
  _Float16* wkH  = (_Float16*)carve((size_t)C_MODEL * C_MODEL * 2);
  _Float16* wvH  = (_Float16*)carve((size_t)C_MODEL * C_MODEL * 2);
  _Float16* woH  = (_Float16*)carve((size_t)C_MODEL * C_MODEL * 2);
  _Float16* w1H  = (_Float16*)carve((size_t)D_FF * C_MODEL * 2);
  _Float16* w2H  = (_Float16*)carve((size_t)C_MODEL * D_FF * 2);
  float*    qF   = (float*)   carve((size_t)BL * C_MODEL * 4);
  float*    kF   = (float*)   carve((size_t)BL * C_MODEL * 4);
  float*    vF   = (float*)   carve((size_t)BL * C_MODEL * 4);
  _Float16* ctxH = (_Float16*)carve((size_t)BL * C_MODEL * 2);
  float*    aoF  = (float*)   carve((size_t)BL * C_MODEL * 4);
  float*    x1F  = (float*)   carve((size_t)BL * C_MODEL * 4);
  _Float16* x1H  = (_Float16*)carve((size_t)BL * C_MODEL * 2);
  _Float16* y1H  = (_Float16*)carve((size_t)BL * D_FF * 2);
  float*    y2F  = (float*)   carve((size_t)BL * C_MODEL * 4);
  float*    x2F  = (float*)   carve((size_t)BL * C_MODEL * 4);
  int*      idx  = (int*)     carve((size_t)L_SEQ * SAMPLEK * 4);

  // sampling indices + f16 weight casts + h
  informer_gen_idx_kernel<<<(L_SEQ * SAMPLEK + 255) / 256, 256, 0, stream>>>(idx);
  informer_f32_to_f16_kernel<<<64, 256, 0, stream>>>(wq, wqH, C_MODEL * C_MODEL);
  informer_f32_to_f16_kernel<<<64, 256, 0, stream>>>(wk, wkH, C_MODEL * C_MODEL);
  informer_f32_to_f16_kernel<<<64, 256, 0, stream>>>(wv, wvH, C_MODEL * C_MODEL);
  informer_f32_to_f16_kernel<<<64, 256, 0, stream>>>(wo, woH, C_MODEL * C_MODEL);
  informer_f32_to_f16_kernel<<<256, 256, 0, stream>>>(w1, w1H, D_FF * C_MODEL);
  informer_f32_to_f16_kernel<<<256, 256, 0, stream>>>(w2, w2H, C_MODEL * D_FF);
  informer_prep_kernel<<<(BL * C_MODEL) / 256, 256, 0, stream>>>(x, mlp_w, mlp_b,
                                                                 hF, hH);

  // QKV projections: [32768,128] @ [128,128]^T, block tile 128x64
  const int blkQKV = (BL / 128) * (C_MODEL / 64);     // 512
  informer_gemm_wmma_kernel<128, false><<<blkQKV, 256, 0, stream>>>(
      hH, wqH, bq, qF, (_Float16*)nullptr, BL, C_MODEL);
  informer_gemm_wmma_kernel<128, false><<<blkQKV, 256, 0, stream>>>(
      hH, wkH, bk, kF, (_Float16*)nullptr, BL, C_MODEL);
  informer_gemm_wmma_kernel<128, false><<<blkQKV, 256, 0, stream>>>(
      hH, wvH, bv, vF, (_Float16*)nullptr, BL, C_MODEL);

  // ProbSparse attention: one block per (seq, head)
  informer_attention_kernel<<<B_SEQ * H_HEADS, L_SEQ, 0, stream>>>(qF, kF, vF,
                                                                   idx, ctxH);

  // output projection
  informer_gemm_wmma_kernel<128, false><<<blkQKV, 256, 0, stream>>>(
      ctxH, woH, bo, aoF, (_Float16*)nullptr, BL, C_MODEL);

  // x1 = LN1(h + attn_out)
  informer_ln_residual_kernel<<<BL, C_MODEL, 0, stream>>>(hF, aoF, ln1g, ln1b,
                                                          x1F, x1H);

  // FFN: relu(x1 @ w1^T + b1) @ w2^T + b2
  const int blkFF1 = (BL / 128) * (D_FF / 64);        // 2048
  informer_gemm_wmma_kernel<128, true><<<blkFF1, 256, 0, stream>>>(
      x1H, w1H, b1, (float*)nullptr, y1H, BL, D_FF);
  informer_gemm_wmma_kernel<512, false><<<blkQKV, 256, 0, stream>>>(
      y1H, w2H, b2, y2F, (_Float16*)nullptr, BL, C_MODEL);

  // x2 = LN2(x1 + y), then sum over N and transpose to [b, C, T]
  informer_ln_residual_kernel<<<BL, C_MODEL, 0, stream>>>(x1F, y2F, ln2g, ln2b,
                                                          x2F, (_Float16*)nullptr);
  informer_final_sum_kernel<<<(4 * C_MODEL * L_SEQ) / 256, 256, 0, stream>>>(
      x2F, out);
}
